// MinRNNCell_2851858285198
// MI455X (gfx1250) — matrix operand order repaired and checked by reading, not compiled, not run on hardware
//
#include <hip/hip_runtime.h>

#define B_DIM 8192
#define IN_DIM 1024
#define H_DIM 2048

#define BM 128
#define BN 128
#define BK 32
#define LDT 40   // LDS tile row stride in ushorts (80 B: 16B-aligned, conflict-friendly)

// Flip to 0 if the async-to-LDS builtin path fails to compile.
#define TRY_ASYNC 1

#if TRY_ASYNC && defined(__has_builtin)
#if __has_builtin(__builtin_amdgcn_global_load_async_to_lds_b128)
#define USE_ASYNC 1
#endif
#endif
#ifndef USE_ASYNC
#define USE_ASYNC 0
#endif

typedef __attribute__((ext_vector_type(16))) __bf16 v16bf;
typedef __attribute__((ext_vector_type(8)))  float  v8f;
typedef __attribute__((ext_vector_type(8)))  unsigned short v8us;

#if USE_ASYNC
// Builtin signature (from hipcc diagnostic): param0 = int4 in addrspace(1) ("__device__"),
// param1 = LDS-side pointer, then imm offset, imm cpol.
typedef int v4i_vs __attribute__((vector_size(4 * sizeof(int))));
typedef __attribute__((address_space(1))) v4i_vs g_v4i;   // global int4
typedef __attribute__((address_space(3))) v4i_vs l_v4i;   // LDS int4
#if __has_builtin(__builtin_amdgcn_s_wait_asynccnt)
#define ASYNC_WAIT(N) __builtin_amdgcn_s_wait_asynccnt(N)
#else
#define ASYNC_WAIT(N) asm volatile("s_wait_asynccnt %0" ::"i"(N) : "memory")
#endif
#else
#define ASYNC_WAIT(N) ((void)0)
#endif

// Copy 16 bytes (8 bf16) global -> LDS.
__device__ __forceinline__ void cp16(unsigned short* lds, const unsigned short* g) {
#if USE_ASYNC
  __builtin_amdgcn_global_load_async_to_lds_b128(
      (g_v4i*)(unsigned short*)g, (l_v4i*)lds, 0, 0);
#else
  *(v8us*)lds = *(const v8us*)g;
#endif
}

__device__ __forceinline__ unsigned short f2bf(float f) {
  unsigned int u = __builtin_bit_cast(unsigned int, f);
  u += 0x7FFFu + ((u >> 16) & 1u);            // round-to-nearest-even
  return (unsigned short)(u >> 16);
}
__device__ __forceinline__ float bf2f(unsigned short s) {
  unsigned int u = ((unsigned int)s) << 16;
  return __builtin_bit_cast(float, u);
}
__device__ __forceinline__ float fast_tanh(float x) {
  float e = __expf(2.0f * x);                  // v_exp_f32
  return 1.0f - 2.0f * __builtin_amdgcn_rcpf(e + 1.0f);
}
__device__ __forceinline__ float fast_sigmoid(float x) {
  return __builtin_amdgcn_rcpf(1.0f + __expf(-x));
}

// Assemble a 16-element bf16 fragment from two 16-byte LDS reads.
__device__ __forceinline__ v16bf frag2(const unsigned short* p0, const unsigned short* p1) {
  union { struct { v8us lo, hi; } s; v16bf v; } r;
  r.s.lo = *(const v8us*)p0;
  r.s.hi = *(const v8us*)p1;
  return r.v;
}

__device__ __forceinline__ v8f wmma_bf16(v16bf a, v16bf b, v8f c) {
  return __builtin_amdgcn_wmma_f32_16x16x32_bf16(false, a, false, b, (short)0, c, false, false);
}

// ---------------------------------------------------------------------------
// fp32 -> bf16 bulk converter (8 elements / thread)
// ---------------------------------------------------------------------------
__global__ __launch_bounds__(256) void cvt_bf16_kernel(const float* __restrict__ src,
                                                       unsigned short* __restrict__ dst) {
  size_t i = ((size_t)blockIdx.x * 256 + threadIdx.x) * 8;
  float4 a = *(const float4*)(src + i);
  float4 b = *(const float4*)(src + i + 4);
  v8us o;
  o[0] = f2bf(a.x); o[1] = f2bf(a.y); o[2] = f2bf(a.z); o[3] = f2bf(a.w);
  o[4] = f2bf(b.x); o[5] = f2bf(b.y); o[6] = f2bf(b.z); o[7] = f2bf(b.w);
  *(v8us*)(dst + i) = o;
}

// ---------------------------------------------------------------------------
// Kernel 1: Zb = bf16( tanh( X @ Wx^T + bx ) )   (bf16 inputs Xb, Wxb)
// ---------------------------------------------------------------------------
__global__ __launch_bounds__(256) void minrnn_z_kernel(
    const unsigned short* __restrict__ Xb, const unsigned short* __restrict__ Wxb,
    const float* __restrict__ bx, unsigned short* __restrict__ Zb)
{
  __shared__ __align__(16) unsigned short sA[2][BM * LDT];
  __shared__ __align__(16) unsigned short sB[2][BN * LDT];

  const int tid   = threadIdx.x;
  const int lane  = tid & 31;
  const int lrow  = lane & 15;
  const int khalf = lane >> 4;
  const int wave  = tid >> 5;
  const int wm    = wave >> 2;     // 0..1 -> 64 rows/wave
  const int wn    = wave & 3;      // 0..3 -> 32 cols/wave
  const int m0    = blockIdx.y * BM;
  const int n0    = blockIdx.x * BN;

  // tile-fill assignment: 2 threads/row, 32 B each
  const int crow  = tid >> 1;
  const int chalf = (tid & 1) * 16;
  const unsigned short* srcA = Xb  + (size_t)(m0 + crow) * IN_DIM + chalf;
  const unsigned short* srcB = Wxb + (size_t)(n0 + crow) * IN_DIM + chalf;

  v8f acc[4][2] = {};

  const int NT = IN_DIM / BK;
#define COPY_TILE1(KT, BI)                                        \
  do {                                                            \
    unsigned short* dA = &sA[BI][crow * LDT + chalf];             \
    unsigned short* dB = &sB[BI][crow * LDT + chalf];             \
    const unsigned short* a = srcA + (KT) * BK;                   \
    const unsigned short* b = srcB + (KT) * BK;                   \
    cp16(dA, a); cp16(dA + 8, a + 8);                             \
    cp16(dB, b); cp16(dB + 8, b + 8);                             \
  } while (0)

  COPY_TILE1(0, 0);
  for (int kt = 0; kt < NT; ++kt) {
    const int cur = kt & 1;
    if (kt + 1 < NT) { COPY_TILE1(kt + 1, cur ^ 1); ASYNC_WAIT(4); }
    else             { ASYNC_WAIT(0); }
    __syncthreads();

    v16bf aF[4], bF[2];
#pragma unroll
    for (int mi = 0; mi < 4; ++mi) {
      const unsigned short* r = &sA[cur][(wm * 64 + mi * 16 + lrow) * LDT];
      aF[mi] = frag2(r + khalf * 8, r + 16 + khalf * 8);
    }
#pragma unroll
    for (int ni = 0; ni < 2; ++ni) {
      const unsigned short* r = &sB[cur][(wn * 32 + ni * 16 + lrow) * LDT];
      bF[ni] = frag2(r + khalf * 8, r + 16 + khalf * 8);
    }
#pragma unroll
    for (int mi = 0; mi < 4; ++mi)
#pragma unroll
      for (int ni = 0; ni < 2; ++ni)
        acc[mi][ni] = wmma_bf16(aF[mi], bF[ni], acc[mi][ni]);
    __syncthreads();
  }

#pragma unroll
  for (int ni = 0; ni < 2; ++ni) {
    int gcol = n0 + wn * 32 + ni * 16 + lrow;
    float bias = bx[gcol];
#pragma unroll
    for (int mi = 0; mi < 4; ++mi) {
#pragma unroll
      for (int r = 0; r < 8; ++r) {
        int grow = m0 + wm * 64 + mi * 16 + khalf * 8 + r;
        Zb[(size_t)grow * H_DIM + gcol] = f2bf(fast_tanh(acc[mi][ni][r] + bias));
      }
    }
  }
}

// ---------------------------------------------------------------------------
// Kernel 2: out = u*h + (1-u)*z,  u = sigmoid(h@Wh^T + z@Uz^T + bu)
// One GEMM with K = 2*H_DIM over bf16 operands (Hb|Zb) x (Whb|Uzb).
// ---------------------------------------------------------------------------
__global__ __launch_bounds__(256) void minrnn_u_kernel(
    const unsigned short* __restrict__ Hb,  const unsigned short* __restrict__ Whb,
    const unsigned short* __restrict__ Uzb, const unsigned short* __restrict__ Zb,
    const float* __restrict__ Hst, const float* __restrict__ bu,
    float* __restrict__ Out)
{
  __shared__ __align__(16) unsigned short sA[2][BM * LDT];
  __shared__ __align__(16) unsigned short sB[2][BN * LDT];

  const int tid   = threadIdx.x;
  const int lane  = tid & 31;
  const int lrow  = lane & 15;
  const int khalf = lane >> 4;
  const int wave  = tid >> 5;
  const int wm    = wave >> 2;
  const int wn    = wave & 3;
  const int m0    = blockIdx.y * BM;
  const int n0    = blockIdx.x * BN;

  const int crow  = tid >> 1;
  const int chalf = (tid & 1) * 16;
  const unsigned short* srcA0 = Hb  + (size_t)(m0 + crow) * H_DIM + chalf;
  const unsigned short* srcA1 = Zb  + (size_t)(m0 + crow) * H_DIM + chalf;
  const unsigned short* srcB0 = Whb + (size_t)(n0 + crow) * H_DIM + chalf;
  const unsigned short* srcB1 = Uzb + (size_t)(n0 + crow) * H_DIM + chalf;

  v8f acc[4][2] = {};

  const int NT  = (2 * H_DIM) / BK;   // 128
  const int NT0 = H_DIM / BK;         // 64, phase boundary (uniform)
#define COPY_TILE2(KT, BI)                                                    \
  do {                                                                        \
    int kt_ = (KT);                                                           \
    unsigned short* dA = &sA[BI][crow * LDT + chalf];                         \
    unsigned short* dB = &sB[BI][crow * LDT + chalf];                         \
    const unsigned short* a;                                                  \
    const unsigned short* b;                                                  \
    if (kt_ < NT0) { a = srcA0 + kt_ * BK;         b = srcB0 + kt_ * BK; }    \
    else           { a = srcA1 + (kt_ - NT0) * BK; b = srcB1 + (kt_ - NT0) * BK; } \
    cp16(dA, a); cp16(dA + 8, a + 8);                                         \
    cp16(dB, b); cp16(dB + 8, b + 8);                                         \
  } while (0)

  COPY_TILE2(0, 0);
  for (int kt = 0; kt < NT; ++kt) {
    const int cur = kt & 1;
    if (kt + 1 < NT) { COPY_TILE2(kt + 1, cur ^ 1); ASYNC_WAIT(4); }
    else             { ASYNC_WAIT(0); }
    __syncthreads();

    v16bf aF[4], bF[2];
#pragma unroll
    for (int mi = 0; mi < 4; ++mi) {
      const unsigned short* r = &sA[cur][(wm * 64 + mi * 16 + lrow) * LDT];
      aF[mi] = frag2(r + khalf * 8, r + 16 + khalf * 8);
    }
#pragma unroll
    for (int ni = 0; ni < 2; ++ni) {
      const unsigned short* r = &sB[cur][(wn * 32 + ni * 16 + lrow) * LDT];
      bF[ni] = frag2(r + khalf * 8, r + 16 + khalf * 8);
    }
#pragma unroll
    for (int mi = 0; mi < 4; ++mi)
#pragma unroll
      for (int ni = 0; ni < 2; ++ni)
        acc[mi][ni] = wmma_bf16(aF[mi], bF[ni], acc[mi][ni]);
    __syncthreads();
  }

#pragma unroll
  for (int ni = 0; ni < 2; ++ni) {
    int gcol = n0 + wn * 32 + ni * 16 + lrow;
    float bias = bu[gcol];
#pragma unroll
    for (int mi = 0; mi < 4; ++mi) {
#pragma unroll
      for (int r = 0; r < 8; ++r) {
        int grow = m0 + wm * 64 + mi * 16 + khalf * 8 + r;
        size_t off = (size_t)grow * H_DIM + gcol;
        float u  = fast_sigmoid(acc[mi][ni][r] + bias);
        float hv = Hst[off];                 // fp32 h for the blend
        float zv = bf2f(Zb[off]);
        Out[off] = u * hv + (1.0f - u) * zv;
      }
    }
  }
}

// ---------------------------------------------------------------------------
extern "C" void kernel_launch(void* const* d_in, const int* in_sizes, int n_in,
                              void* d_out, int out_size, void* d_ws, size_t ws_size,
                              hipStream_t stream) {
  (void)in_sizes; (void)n_in; (void)out_size; (void)ws_size;
  const float* prev = (const float*)d_in[0];   // [8192, 2048]
  const float* x    = (const float*)d_in[1];   // [8192, 1024]
  const float* Wx   = (const float*)d_in[2];   // [2048, 1024]
  const float* bx   = (const float*)d_in[3];   // [2048]
  const float* Wh   = (const float*)d_in[4];   // [2048, 2048]
  const float* Uz   = (const float*)d_in[5];   // [2048, 2048]
  const float* bu   = (const float*)d_in[6];   // [2048]
  float* out = (float*)d_out;

  // Workspace layout (ushort units)
  unsigned short* ws = (unsigned short*)d_ws;
  const size_t nZ  = (size_t)B_DIM * H_DIM;    // 16M
  const size_t nH  = (size_t)B_DIM * H_DIM;    // 16M
  const size_t nX  = (size_t)B_DIM * IN_DIM;   // 8M
  const size_t nWx = (size_t)H_DIM * IN_DIM;   // 2M
  const size_t nWh = (size_t)H_DIM * H_DIM;    // 4M
  unsigned short* zb  = ws;
  unsigned short* hb  = zb  + nZ;
  unsigned short* xb  = hb  + nH;
  unsigned short* wxb = xb  + nX;
  unsigned short* whb = wxb + nWx;
  unsigned short* uzb = whb + nWh;

  dim3 block(256);
  cvt_bf16_kernel<<<dim3((unsigned)(nH  / 2048)), block, 0, stream>>>(prev, hb);
  cvt_bf16_kernel<<<dim3((unsigned)(nX  / 2048)), block, 0, stream>>>(x,    xb);
  cvt_bf16_kernel<<<dim3((unsigned)(nWx / 2048)), block, 0, stream>>>(Wx,   wxb);
  cvt_bf16_kernel<<<dim3((unsigned)(nWh / 2048)), block, 0, stream>>>(Wh,   whb);
  cvt_bf16_kernel<<<dim3((unsigned)(nWh / 2048)), block, 0, stream>>>(Uz,   uzb);

  dim3 grid(H_DIM / BN, B_DIM / BM);           // (16, 64)
  minrnn_z_kernel<<<grid, block, 0, stream>>>(xb, wxb, bx, zb);
  minrnn_u_kernel<<<grid, block, 0, stream>>>(hb, whb, uzb, zb, prev, bu, out);
}
